// Transformer_62397284876907
// MI455X (gfx1250) — matrix operand order repaired and checked by reading, not compiled, not run on hardware
//
#include <hip/hip_runtime.h>
#include <hip/hip_bf16.h>

// ---------------------------------------------------------------------------
// Model dimensions (compile-time, from the reference)
// ---------------------------------------------------------------------------
#define D_MODEL 512
#define N_HEADS 8
#define D_HEAD  64
#define DFF_DIM 2048
#define SEQ     1024
#define BATCH   2
#define NEL     4
#define NDL     4
#define VOCT    32000

typedef __attribute__((ext_vector_type(16))) __bf16 v16bf;
typedef __attribute__((ext_vector_type(8)))  float  v8f;
typedef __attribute__((ext_vector_type(4)))  unsigned v4u;
typedef __attribute__((ext_vector_type(8)))  int      v8i;
typedef __attribute__((ext_vector_type(4)))  int      v4i;

union AFrag { v16bf v; unsigned u[8]; };

static __device__ __forceinline__ v8f v8f_zero() {
  v8f z = {0.f, 0.f, 0.f, 0.f, 0.f, 0.f, 0.f, 0.f};
  return z;
}

// ---------------------------------------------------------------------------
// Tensor Data Mover (gfx1250): DMA a 2D bf16 tile (rows x cols, row stride
// `strideElems`) from global memory into LDS at byte offset `ldsOff`.
// Descriptor layout per CDNA5 ISA 8.3/8.4 (groups 0/1; groups 2/3 size-1).
// This toolchain exposes the 6-argument builtin form:
//   (uint32x4 g0, int32x8 g1, int32x4 g2, int32x4 g3, int32x8 g4, i32 cpol)
// Guarded: falls back to vector-load staging if the builtin is unavailable.
// ---------------------------------------------------------------------------
#if __has_builtin(__builtin_amdgcn_tensor_load_to_lds) && \
    __has_builtin(__builtin_amdgcn_s_wait_tensorcnt)
#define HAS_TDM 1
#else
#define HAS_TDM 0
#endif

#if HAS_TDM
static __device__ __forceinline__ void tdm_load_tile_2d(
    const __hip_bfloat16* gptr, unsigned ldsOff, unsigned strideElems,
    unsigned rows, unsigned cols) {
  const unsigned long long ga = (unsigned long long)(size_t)gptr;
  v4u g0;
  g0.x = 1u;                                        // count=1, user mode
  g0.y = ldsOff;                                    // lds_addr (bytes)
  g0.z = (unsigned)(ga & 0xffffffffu);              // global_addr[31:0]
  g0.w = (unsigned)((ga >> 32) & 0x01ffffffu) | (2u << 30);  // [56:32] | type=2
  const unsigned td0 = strideElems;                 // tensor_dim0 (elems/row)
  const unsigned td1 = 0x7fffffffu;                 // tensor_dim1: huge (no OOB)
  const unsigned long long st0 = (unsigned long long)strideElems;
  v8i g1;
  g1.s0 = 0x00010000;                               // data_size=1 (2B), no flags
  g1.s1 = (int)((td0 & 0xffffu) << 16);             // ab_addr=0 | td0[15:0]
  g1.s2 = (int)(((td0 >> 16) & 0xffffu) | ((td1 & 0xffffu) << 16));
  g1.s3 = (int)(((td1 >> 16) & 0xffffu) | ((cols & 0xffffu) << 16)); // tile_dim0
  g1.s4 = (int)(rows & 0xffffu);                    // tile_dim1 | tile_dim2=0
  g1.s5 = (int)(st0 & 0xffffffffu);                 // tensor_dim0_stride lo
  g1.s6 = (int)((st0 >> 32) & 0xffffu);             // stride hi | dim1_stride=0
  g1.s7 = 0;
  v4i g2; g2.x = 1; g2.y = 1; g2.z = 0; g2.w = 0;   // dims 2/3 size-1, tile_dim3=0
  v4i g3; g3.x = 0; g3.y = 0; g3.z = 0; g3.w = 0;   // tile_dim4=0
  v8i g4; g4.s0 = 0; g4.s1 = 0; g4.s2 = 0; g4.s3 = 0;
          g4.s4 = 0; g4.s5 = 0; g4.s6 = 0; g4.s7 = 0;   // unused extra group
  __builtin_amdgcn_tensor_load_to_lds(g0, g1, g2, g3, g4, 0);
}
#endif

static __device__ __forceinline__ unsigned lds_byte_offset(const void* p) {
  return (unsigned)(size_t)p;   // low 32 bits of the LDS aperture address
}

// ---------------------------------------------------------------------------
// f32 -> bf16 conversion (weights)
// ---------------------------------------------------------------------------
__global__ void f2bf_kernel(const float* __restrict__ s,
                            __hip_bfloat16* __restrict__ d, size_t n) {
  size_t i = (size_t)blockIdx.x * blockDim.x + threadIdx.x;
  const size_t stride = (size_t)gridDim.x * blockDim.x;
  for (; i < n; i += stride) d[i] = __float2bfloat16(s[i]);
}

// ---------------------------------------------------------------------------
// Embedding + sinusoidal positional encoding
// ---------------------------------------------------------------------------
__global__ __launch_bounds__(256) void embed_pe_kernel(
    const int* __restrict__ tok, const float* __restrict__ emb,
    float* __restrict__ x, __hip_bfloat16* __restrict__ xb) {
  const int rs = blockIdx.x;            // b*SEQ + s
  const int s  = rs & (SEQ - 1);
  const int t  = threadIdx.x;
  const int id = tok[rs];
  const float div = __expf((float)(2 * t) * (-9.210340371976184f / (float)D_MODEL));
  const float ang = (float)s * div;
  const size_t base = (size_t)rs * D_MODEL;
  const float e0 = emb[(size_t)id * D_MODEL + 2 * t]     + __sinf(ang);
  const float e1 = emb[(size_t)id * D_MODEL + 2 * t + 1] + __cosf(ang);
  x[base + 2 * t]     = e0;
  x[base + 2 * t + 1] = e1;
  xb[base + 2 * t]     = __float2bfloat16(e0);
  xb[base + 2 * t + 1] = __float2bfloat16(e1);
}

// ---------------------------------------------------------------------------
// Residual add + LayerNorm over D=512; writes f32 + bf16 copies
// ---------------------------------------------------------------------------
__global__ __launch_bounds__(256) void add_ln_kernel(
    const float* __restrict__ x, const float* __restrict__ a,
    const float* __restrict__ g, const float* __restrict__ be,
    float* __restrict__ y, __hip_bfloat16* __restrict__ yb) {
  __shared__ float red[256];
  const int row = blockIdx.x;
  const int t = threadIdx.x;
  const size_t base = (size_t)row * D_MODEL;
  const float v0 = x[base + t]       + a[base + t];
  const float v1 = x[base + t + 256] + a[base + t + 256];
  red[t] = v0 + v1;
  __syncthreads();
  for (int sft = 128; sft > 0; sft >>= 1) {
    if (t < sft) red[t] += red[t + sft];
    __syncthreads();
  }
  const float mean = red[0] * (1.0f / (float)D_MODEL);
  __syncthreads();
  const float d0 = v0 - mean, d1 = v1 - mean;
  red[t] = d0 * d0 + d1 * d1;
  __syncthreads();
  for (int sft = 128; sft > 0; sft >>= 1) {
    if (t < sft) red[t] += red[t + sft];
    __syncthreads();
  }
  const float inv = rsqrtf(red[0] * (1.0f / (float)D_MODEL) + 1e-5f);
  const float o0 = d0 * inv * g[t]       + be[t];
  const float o1 = d1 * inv * g[t + 256] + be[t + 256];
  y[base + t]       = o0;
  y[base + t + 256] = o1;
  yb[base + t]       = __float2bfloat16(o0);
  yb[base + t + 256] = __float2bfloat16(o1);
}

// ---------------------------------------------------------------------------
// bf16 GEMM, f32 accumulate, v_wmma_f32_16x16x32_bf16.
//   C[M,N] = A[M,K] bf16 row-major x B[K,N] bf16 row-major + bias
// Block tile 128x64, BK=32, 256 threads = 8 waves (4x2), 32x32 per wave.
// Double-buffered LDS (one barrier/iter); A tile DMA'd by the Tensor Data
// Mover (wave 0) when available, B tile staged transposed by vector loads.
// Next tile's loads are issued before the current tile's WMMAs.
// ---------------------------------------------------------------------------
template <bool RELU, bool OUTF32, bool OUTBF>
__global__ __launch_bounds__(256) void gemm_bf16_kernel(
    const __hip_bfloat16* __restrict__ A, const __hip_bfloat16* __restrict__ B,
    const float* __restrict__ bias, float* __restrict__ Cf,
    __hip_bfloat16* __restrict__ Cb, int M, int N, int K) {
  __shared__ __hip_bfloat16 As[2][128][32];
  __shared__ __hip_bfloat16 Bt[2][64][32];   // transposed: [n][k]
  const int tid  = threadIdx.x;
  const int n0   = blockIdx.x * 64;
  const int m0   = blockIdx.y * 128;
  const int lane = tid & 31, wave = tid >> 5;
  const int lm   = lane & 15, lh = lane >> 4;
  const int wr   = wave >> 1, wc = wave & 1;   // 4 x 2 wave grid

  v8f acc[2][2];
  acc[0][0] = v8f_zero(); acc[0][1] = v8f_zero();
  acc[1][0] = v8f_zero(); acc[1][1] = v8f_zero();

  const int ar  = tid >> 1, ac = (tid & 1) * 16;   // A staging: 128 x 32
  const int bkr = tid >> 3, bc0 = (tid & 7) * 8;   // B staging: 32 x 64

  uint4 bregs;
#if !HAS_TDM
  uint4 aregs0, aregs1;
#endif

  // ---- prologue: start tile 0 ----
#if HAS_TDM
  if (wave == 0)
    tdm_load_tile_2d(A + (size_t)m0 * K, lds_byte_offset(&As[0][0][0]),
                     (unsigned)K, 128, 32);
#else
  {
    const uint4* ap = reinterpret_cast<const uint4*>(A + (size_t)(m0 + ar) * K + ac);
    aregs0 = ap[0];
    aregs1 = ap[1];
  }
#endif
  bregs = *reinterpret_cast<const uint4*>(B + (size_t)bkr * N + n0 + bc0);

  int cur = 0;
  for (int k0 = 0; k0 < K; k0 += 32) {
    // --- commit staged tile into LDS[cur] ---
#if !HAS_TDM
    *reinterpret_cast<uint4*>(&As[cur][ar][ac])     = aregs0;
    *reinterpret_cast<uint4*>(&As[cur][ar][ac + 8]) = aregs1;
#endif
    {
      const unsigned short* bs = reinterpret_cast<const unsigned short*>(&bregs);
#pragma unroll
      for (int j = 0; j < 8; ++j)
        *reinterpret_cast<unsigned short*>(&Bt[cur][bc0 + j][bkr]) = bs[j];
    }
#if HAS_TDM
    if (wave == 0) __builtin_amdgcn_s_wait_tensorcnt(0);
#endif
    __syncthreads();

    // --- issue next tile's loads (land in LDS[cur^1] / regs) ---
    if (k0 + 32 < K) {
#if HAS_TDM
      if (wave == 0)
        tdm_load_tile_2d(A + (size_t)m0 * K + (k0 + 32),
                         lds_byte_offset(&As[cur ^ 1][0][0]), (unsigned)K, 128,
                         32);
#else
      const uint4* ap =
          reinterpret_cast<const uint4*>(A + (size_t)(m0 + ar) * K + k0 + 32 + ac);
      aregs0 = ap[0];
      aregs1 = ap[1];
#endif
      bregs = *reinterpret_cast<const uint4*>(B + (size_t)(k0 + 32 + bkr) * N +
                                              n0 + bc0);
      __builtin_prefetch(B + (size_t)(k0 + 64 + bkr) * N + n0 + bc0, 0, 1);
    }

    // --- fragments + WMMA from LDS[cur] ---
    AFrag af[2], bf[2];
#pragma unroll
    for (int t = 0; t < 2; ++t) {
      const int row = wr * 32 + t * 16 + lm;
#pragma unroll
      for (int j = 0; j < 8; ++j) {          // A layout: half0 -> K 0-15
        const int kk = 2 * j + 16 * lh;
        af[t].u[j] = *reinterpret_cast<const unsigned*>(&As[cur][row][kk]);
      }
      const int col = wc * 32 + t * 16 + lm;
#pragma unroll
      for (int j = 0; j < 8; ++j) {          // B layout: half0 -> K 0-15
        const int kk = 2 * j + 16 * lh;
        bf[t].u[j] = *reinterpret_cast<const unsigned*>(&Bt[cur][col][kk]);
      }
    }
#pragma unroll
    for (int tr = 0; tr < 2; ++tr)
#pragma unroll
      for (int tc = 0; tc < 2; ++tc)
        acc[tr][tc] = __builtin_amdgcn_wmma_f32_16x16x32_bf16(
            false, af[tr].v, false, bf[tc].v, (short)0, acc[tr][tc], false,
            false);
    cur ^= 1;
  }

  // --- epilogue: C layout VGPR r -> M = r + 8*half, N = lane&15 ---
#pragma unroll
  for (int tr = 0; tr < 2; ++tr)
#pragma unroll
    for (int tc = 0; tc < 2; ++tc) {
      const int n = n0 + wc * 32 + tc * 16 + lm;
      const float bz = bias[n];
#pragma unroll
      for (int r = 0; r < 8; ++r) {
        const int m = m0 + wr * 32 + tr * 16 + r + 8 * lh;
        float v = acc[tr][tc][r] + bz;
        if (RELU) v = fmaxf(v, 0.f);
        const size_t idx = (size_t)m * N + n;
        if (OUTF32) Cf[idx] = v;
        if (OUTBF)  Cb[idx] = __float2bfloat16(v);
      }
    }
}

// ---------------------------------------------------------------------------
// Fused attention: one workgroup per (b, h, 16-query tile), 128 threads
// (4 waves). Phase 1: S = Q K^T / 8 via WMMA, K-fragments double-buffered in
// registers, with branchless mask. Phase 2: softmax over the 1024-key row in LDS.
// Phase 3: O = P V via WMMA with a double-buffered V-transpose LDS tile.
// ---------------------------------------------------------------------------
#define ATTN_SMEM_BYTES \
  (16 * SEQ * 4 + 16 * SEQ * 2 + 2 * D_HEAD * 32 * 2 + 2 * 16 * 8 * 4)

template <bool MASKED>
__global__ __launch_bounds__(128) void attention_kernel(
    const __hip_bfloat16* __restrict__ Q, const __hip_bfloat16* __restrict__ Km,
    const __hip_bfloat16* __restrict__ Vm, __hip_bfloat16* __restrict__ O,
    const int* __restrict__ tok) {
  extern __shared__ char smem[];
  float* Sbuf = reinterpret_cast<float*>(smem);                      // 16 x SEQ
  __hip_bfloat16* Pbuf =
      reinterpret_cast<__hip_bfloat16*>(smem + 16 * SEQ * 4);        // 16 x SEQ
  __hip_bfloat16* Vt = Pbuf + 16 * SEQ;                              // 2x[64][32]
  float* redm = reinterpret_cast<float*>(smem + 16 * SEQ * 4 + 16 * SEQ * 2 +
                                         2 * D_HEAD * 32 * 2);
  float* reds = redm + 16 * 8;

  const int qt = blockIdx.x, h = blockIdx.y, b = blockIdx.z;
  const int tid = threadIdx.x;
  const int lane = tid & 31, wave = tid >> 5;
  const int lm = lane & 15, lh = lane >> 4;

  // --- Q fragments (16 x 64, two 32-wide chunks), reused across all tiles ---
  AFrag qf[2];
  {
    const size_t qbase =
        ((size_t)(b * SEQ + qt * 16 + lm)) * D_MODEL + h * D_HEAD;
#pragma unroll
    for (int c = 0; c < 2; ++c)
#pragma unroll
      for (int j = 0; j < 8; ++j) {
        const int kk = 2 * j + 16 * lh + 32 * c;
        qf[c].u[j] = *reinterpret_cast<const unsigned*>(Q + qbase + kk);
      }
  }

  // --- hoisted per-row mask info (8 token loads total per lane) ---
  bool tokok[8];
  if (MASKED) {
#pragma unroll
    for (int r = 0; r < 8; ++r)
      tokok[r] = tok[b * SEQ + qt * 16 + r + 8 * lh] != 0;
  }

  // --- Phase 1: scores; each wave owns 16 of the 64 key tiles, K-fragments
  // double-buffered so next tile's loads overlap current WMMAs ---
  const int kt0 = wave * 16, ktEnd = kt0 + 16;
  AFrag kf[2][2];
  {
    const size_t kbase =
        ((size_t)(b * SEQ + kt0 * 16 + lm)) * D_MODEL + h * D_HEAD;
#pragma unroll
    for (int c = 0; c < 2; ++c)
#pragma unroll
      for (int j = 0; j < 8; ++j)
        kf[0][c].u[j] = *reinterpret_cast<const unsigned*>(
            Km + kbase + 2 * j + 16 * lh + 32 * c);
  }
  int pb = 0;
  for (int kt = kt0; kt < ktEnd; ++kt) {
    if (kt + 1 < ktEnd) {
      const size_t kbase =
          ((size_t)(b * SEQ + (kt + 1) * 16 + lm)) * D_MODEL + h * D_HEAD;
#pragma unroll
      for (int c = 0; c < 2; ++c)
#pragma unroll
        for (int j = 0; j < 8; ++j)
          kf[pb ^ 1][c].u[j] = *reinterpret_cast<const unsigned*>(
              Km + kbase + 2 * j + 16 * lh + 32 * c);
    }
    v8f sf = v8f_zero();
    sf = __builtin_amdgcn_wmma_f32_16x16x32_bf16(false, qf[0].v, false,
                                                 kf[pb][0].v, (short)0, sf,
                                                 false, false);
    sf = __builtin_amdgcn_wmma_f32_16x16x32_bf16(false, qf[1].v, false,
                                                 kf[pb][1].v, (short)0, sf,
                                                 false, false);
#pragma unroll
    for (int r = 0; r < 8; ++r) {
      const int row = r + 8 * lh;
      const int col = kt * 16 + lm;
      float sv = sf[r] * 0.125f;            // 1/sqrt(64)
      if (MASKED) {
        const bool ok = (col <= qt * 16 + row) && tokok[r];
        sv = ok ? sv : -1.0e9f;             // branchless v_cndmask
      }
      Sbuf[row * SEQ + col] = sv;
    }
    pb ^= 1;
  }
  __syncthreads();

  // --- Phase 2: softmax, 8 threads per row ---
  const int sr = tid >> 3;
  const int sc = tid & 7;
  float mx = -3.4e38f;
  for (int c = sc; c < SEQ; c += 8) mx = fmaxf(mx, Sbuf[sr * SEQ + c]);
  redm[sr * 8 + sc] = mx;
  __syncthreads();
  if (sc == 0) {
    float m2 = redm[sr * 8];
    for (int j = 1; j < 8; ++j) m2 = fmaxf(m2, redm[sr * 8 + j]);
    redm[sr * 8] = m2;
  }
  __syncthreads();
  const float rmax = redm[sr * 8];
  float sum = 0.f;
  for (int c = sc; c < SEQ; c += 8) {
    const float e = __expf(Sbuf[sr * SEQ + c] - rmax);
    Sbuf[sr * SEQ + c] = e;
    sum += e;
  }
  reds[sr * 8 + sc] = sum;
  __syncthreads();
  if (sc == 0) {
    float s2 = 0.f;
    for (int j = 0; j < 8; ++j) s2 += reds[sr * 8 + j];
    reds[sr * 8] = s2;
  }
  __syncthreads();
  const float inv = 1.0f / reds[sr * 8];
  for (int c = sc; c < SEQ; c += 8)
    Pbuf[sr * SEQ + c] = __float2bfloat16(Sbuf[sr * SEQ + c] * inv);
  __syncthreads();

  // --- Phase 3: O = P V; wave w owns head-dim cols [16w,16w+16);
  // V chunk staged transposed, double-buffered (one barrier/iter) ---
  v8f of = v8f_zero();
  const int d0 = wave * 16;
  const int kl = tid >> 2;                  // key local 0..31
  const int dg = (tid & 3) * 16;            // d group
  uint4 v0r, v1r;
  {
    const uint4* vp = reinterpret_cast<const uint4*>(
        Vm + ((size_t)(b * SEQ + kl)) * D_MODEL + h * D_HEAD + dg);
    v0r = vp[0];
    v1r = vp[1];
  }
  int vc = 0;
  for (int kk0 = 0; kk0 < SEQ; kk0 += 32) {
    {  // scatter staged regs into Vt[vc][d][key]
      __hip_bfloat16* vt = Vt + vc * (D_HEAD * 32);
      const unsigned short* vs0 = reinterpret_cast<const unsigned short*>(&v0r);
      const unsigned short* vs1 = reinterpret_cast<const unsigned short*>(&v1r);
#pragma unroll
      for (int j = 0; j < 8; ++j) {
        *reinterpret_cast<unsigned short*>(&vt[(dg + j) * 32 + kl])     = vs0[j];
        *reinterpret_cast<unsigned short*>(&vt[(dg + 8 + j) * 32 + kl]) = vs1[j];
      }
    }
    __syncthreads();
    if (kk0 + 32 < SEQ) {
      const uint4* vp = reinterpret_cast<const uint4*>(
          Vm + ((size_t)(b * SEQ + kk0 + 32 + kl)) * D_MODEL + h * D_HEAD + dg);
      v0r = vp[0];
      v1r = vp[1];
    }
    const __hip_bfloat16* vt = Vt + vc * (D_HEAD * 32);
    AFrag pf, vf;
#pragma unroll
    for (int j = 0; j < 8; ++j) {
      const int kk = 2 * j + 16 * lh;
      pf.u[j] = *reinterpret_cast<const unsigned*>(&Pbuf[lm * SEQ + kk0 + kk]);
    }
#pragma unroll
    for (int j = 0; j < 8; ++j) {
      const int kk = 2 * j + 16 * lh;
      vf.u[j] = *reinterpret_cast<const unsigned*>(&vt[(d0 + lm) * 32 + kk]);
    }
    of = __builtin_amdgcn_wmma_f32_16x16x32_bf16(false, pf.v, false, vf.v,
                                                 (short)0, of, false, false);
    vc ^= 1;
  }
#pragma unroll
  for (int r = 0; r < 8; ++r) {
    const int row = qt * 16 + r + 8 * lh;
    O[((size_t)(b * SEQ + row)) * D_MODEL + h * D_HEAD + d0 + lm] =
        __float2bfloat16(of[r]);
  }
}

// ---------------------------------------------------------------------------
// Host-side orchestration
// ---------------------------------------------------------------------------
extern "C" void kernel_launch(void* const* d_in, const int* in_sizes, int n_in,
                              void* d_out, int out_size, void* d_ws,
                              size_t ws_size, hipStream_t stream) {
  (void)in_sizes; (void)n_in; (void)out_size; (void)ws_size;
  typedef __hip_bfloat16 bf;

  const int*   src        = (const int*)d_in[0];
  const int*   tgt        = (const int*)d_in[1];
  const float* emb_enc    = (const float*)d_in[2];
  const float* emb_dec    = (const float*)d_in[3];
  const float* enc_attn_w = (const float*)d_in[4];
  const float* enc_attn_b = (const float*)d_in[5];
  const float* enc_ffn_w1 = (const float*)d_in[6];
  const float* enc_ffn_b1 = (const float*)d_in[7];
  const float* enc_ffn_w2 = (const float*)d_in[8];
  const float* enc_ffn_b2 = (const float*)d_in[9];
  const float* enc_ln_g   = (const float*)d_in[10];
  const float* enc_ln_b   = (const float*)d_in[11];
  const float* dec_self_w = (const float*)d_in[12];
  const float* dec_self_b = (const float*)d_in[13];
  const float* dec_cross_w= (const float*)d_in[14];
  const float* dec_cross_b= (const float*)d_in[15];
  const float* dec_ffn_w1 = (const float*)d_in[16];
  const float* dec_ffn_b1 = (const float*)d_in[17];
  const float* dec_ffn_w2 = (const float*)d_in[18];
  const float* dec_ffn_b2 = (const float*)d_in[19];
  const float* dec_ln_g   = (const float*)d_in[20];
  const float* dec_ln_b   = (const float*)d_in[21];
  const float* fc_w       = (const float*)d_in[22];
  const float* fc_b       = (const float*)d_in[23];
  float* out = (float*)d_out;

  const size_t M   = (size_t)BATCH * SEQ;              // 2048 rows
  const size_t DD  = (size_t)D_MODEL * D_MODEL;        // 262144
  const size_t DF  = (size_t)D_MODEL * DFF_DIM;        // 1048576

  char* ws = (char*)d_ws;
  size_t off = 0;
  auto alloc = [&](size_t bytes) -> char* {
    off = (off + 255) & ~(size_t)255;
    char* p = ws + off;
    off += bytes;
    return p;
  };

  bf* w_enc_attn  = (bf*)alloc(NEL * 4 * DD * 2);
  bf* w_enc_f1    = (bf*)alloc(NEL * DF * 2);
  bf* w_enc_f2    = (bf*)alloc(NEL * DF * 2);
  bf* w_dec_self  = (bf*)alloc(NDL * 4 * DD * 2);
  bf* w_dec_cross = (bf*)alloc(NDL * 4 * DD * 2);
  bf* w_dec_f1    = (bf*)alloc(NDL * DF * 2);
  bf* w_dec_f2    = (bf*)alloc(NDL * DF * 2);
  bf* w_fc        = (bf*)alloc((size_t)D_MODEL * VOCT * 2);

  float* xeA   = (float*)alloc(M * D_MODEL * 4);
  float* xeB   = (float*)alloc(M * D_MODEL * 4);
  bf*    xe_bf = (bf*)alloc(M * D_MODEL * 2);
  float* xdA   = (float*)alloc(M * D_MODEL * 4);
  float* xdB   = (float*)alloc(M * D_MODEL * 4);
  bf*    xd_bf = (bf*)alloc(M * D_MODEL * 2);
  bf*    qb    = (bf*)alloc(M * D_MODEL * 2);
  bf*    kb    = (bf*)alloc(M * D_MODEL * 2);
  bf*    vb    = (bf*)alloc(M * D_MODEL * 2);
  bf*    ob    = (bf*)alloc(M * D_MODEL * 2);
  float* tmp   = (float*)alloc(M * D_MODEL * 4);
  bf*    hid   = (bf*)alloc(M * DFF_DIM * 2);

  auto conv = [&](const float* s, bf* d, size_t n) {
    f2bf_kernel<<<dim3(2048), dim3(256), 0, stream>>>(s, d, n);
  };
  conv(enc_attn_w,  w_enc_attn,  NEL * 4 * DD);
  conv(enc_ffn_w1,  w_enc_f1,    NEL * DF);
  conv(enc_ffn_w2,  w_enc_f2,    NEL * DF);
  conv(dec_self_w,  w_dec_self,  NDL * 4 * DD);
  conv(dec_cross_w, w_dec_cross, NDL * 4 * DD);
  conv(dec_ffn_w1,  w_dec_f1,    NDL * DF);
  conv(dec_ffn_w2,  w_dec_f2,    NDL * DF);
  conv(fc_w,        w_fc,        (size_t)D_MODEL * VOCT);

  const unsigned mtiles = (unsigned)(M / 128);
  auto gemm_bf = [&](const bf* A, const bf* W, const float* bias, bf* Ob,
                     int N, int K) {   // bias, bf16 output (QKV proj)
    gemm_bf16_kernel<false, false, true>
        <<<dim3((unsigned)(N / 64), mtiles), 256, 0, stream>>>(
            A, W, bias, nullptr, Ob, (int)M, N, K);
  };
  auto gemm_f32 = [&](const bf* A, const bf* W, const float* bias, float* Of,
                      int N, int K) {  // bias, f32 output (O-proj/FFN2/FC)
    gemm_bf16_kernel<false, true, false>
        <<<dim3((unsigned)(N / 64), mtiles), 256, 0, stream>>>(
            A, W, bias, Of, nullptr, (int)M, N, K);
  };
  auto gemm_relu = [&](const bf* A, const bf* W, const float* bias, bf* Ob,
                       int N, int K) { // bias+ReLU, bf16 output (FFN1)
    gemm_bf16_kernel<true, false, true>
        <<<dim3((unsigned)(N / 64), mtiles), 256, 0, stream>>>(
            A, W, bias, nullptr, Ob, (int)M, N, K);
  };
  auto attn = [&](const bf* Qp, const bf* Kp, const bf* Vp, bf* Op,
                  const int* tokp, bool masked) {
    if (masked)
      attention_kernel<true><<<dim3(SEQ / 16, N_HEADS, BATCH), 128,
                               ATTN_SMEM_BYTES, stream>>>(Qp, Kp, Vp, Op, tokp);
    else
      attention_kernel<false><<<dim3(SEQ / 16, N_HEADS, BATCH), 128,
                                ATTN_SMEM_BYTES, stream>>>(Qp, Kp, Vp, Op, tokp);
  };
  auto lnorm = [&](const float* x, const float* a, const float* g,
                   const float* be, float* y, bf* yb) {
    add_ln_kernel<<<dim3((unsigned)M), 256, 0, stream>>>(x, a, g, be, y, yb);
  };

  // Embedding + PE
  embed_pe_kernel<<<dim3((unsigned)M), 256, 0, stream>>>(src, emb_enc, xeA, xe_bf);
  embed_pe_kernel<<<dim3((unsigned)M), 256, 0, stream>>>(tgt, emb_dec, xdA, xd_bf);

  // ---- Encoder ----
  float* ec = xeA; float* en = xeB;
  for (int i = 0; i < NEL; ++i) {
    const bf* W = w_enc_attn + (size_t)i * 4 * DD;
    const float* Bb = enc_attn_b + (size_t)i * 4 * D_MODEL;
    gemm_bf(xe_bf, W + 0 * DD, Bb + 0 * D_MODEL, qb, D_MODEL, D_MODEL);
    gemm_bf(xe_bf, W + 1 * DD, Bb + 1 * D_MODEL, kb, D_MODEL, D_MODEL);
    gemm_bf(xe_bf, W + 2 * DD, Bb + 2 * D_MODEL, vb, D_MODEL, D_MODEL);
    attn(qb, kb, vb, ob, nullptr, false);
    gemm_f32(ob, W + 3 * DD, Bb + 3 * D_MODEL, tmp, D_MODEL, D_MODEL);
    lnorm(ec, tmp, enc_ln_g + (size_t)(i * 2 + 0) * D_MODEL,
          enc_ln_b + (size_t)(i * 2 + 0) * D_MODEL, en, xe_bf);
    { float* t = ec; ec = en; en = t; }
    gemm_relu(xe_bf, w_enc_f1 + (size_t)i * DF, enc_ffn_b1 + (size_t)i * DFF_DIM,
              hid, DFF_DIM, D_MODEL);
    gemm_f32(hid, w_enc_f2 + (size_t)i * DF, enc_ffn_b2 + (size_t)i * D_MODEL,
             tmp, D_MODEL, DFF_DIM);
    lnorm(ec, tmp, enc_ln_g + (size_t)(i * 2 + 1) * D_MODEL,
          enc_ln_b + (size_t)(i * 2 + 1) * D_MODEL, en, xe_bf);
    { float* t = ec; ec = en; en = t; }
  }
  // xe_bf now holds the final encoder output (bf16) for cross-attention.

  // ---- Decoder ----
  float* dc = xdA; float* dn = xdB;
  for (int i = 0; i < NDL; ++i) {
    // masked self-attention
    const bf* Ws = w_dec_self + (size_t)i * 4 * DD;
    const float* Bs = dec_self_b + (size_t)i * 4 * D_MODEL;
    gemm_bf(xd_bf, Ws + 0 * DD, Bs + 0 * D_MODEL, qb, D_MODEL, D_MODEL);
    gemm_bf(xd_bf, Ws + 1 * DD, Bs + 1 * D_MODEL, kb, D_MODEL, D_MODEL);
    gemm_bf(xd_bf, Ws + 2 * DD, Bs + 2 * D_MODEL, vb, D_MODEL, D_MODEL);
    attn(qb, kb, vb, ob, tgt, true);
    gemm_f32(ob, Ws + 3 * DD, Bs + 3 * D_MODEL, tmp, D_MODEL, D_MODEL);
    lnorm(dc, tmp, dec_ln_g + (size_t)(i * 3 + 0) * D_MODEL,
          dec_ln_b + (size_t)(i * 3 + 0) * D_MODEL, dn, xd_bf);
    { float* t = dc; dc = dn; dn = t; }
    // cross-attention (K/V from encoder output)
    const bf* Wc = w_dec_cross + (size_t)i * 4 * DD;
    const float* Bc = dec_cross_b + (size_t)i * 4 * D_MODEL;
    gemm_bf(xd_bf, Wc + 0 * DD, Bc + 0 * D_MODEL, qb, D_MODEL, D_MODEL);
    gemm_bf(xe_bf, Wc + 1 * DD, Bc + 1 * D_MODEL, kb, D_MODEL, D_MODEL);
    gemm_bf(xe_bf, Wc + 2 * DD, Bc + 2 * D_MODEL, vb, D_MODEL, D_MODEL);
    attn(qb, kb, vb, ob, nullptr, false);
    gemm_f32(ob, Wc + 3 * DD, Bc + 3 * D_MODEL, tmp, D_MODEL, D_MODEL);
    lnorm(dc, tmp, dec_ln_g + (size_t)(i * 3 + 1) * D_MODEL,
          dec_ln_b + (size_t)(i * 3 + 1) * D_MODEL, dn, xd_bf);
    { float* t = dc; dc = dn; dn = t; }
    // FFN
    gemm_relu(xd_bf, w_dec_f1 + (size_t)i * DF, dec_ffn_b1 + (size_t)i * DFF_DIM,
              hid, DFF_DIM, D_MODEL);
    gemm_f32(hid, w_dec_f2 + (size_t)i * DF, dec_ffn_b2 + (size_t)i * D_MODEL,
             tmp, D_MODEL, DFF_DIM);
    lnorm(dc, tmp, dec_ln_g + (size_t)(i * 3 + 2) * D_MODEL,
          dec_ln_b + (size_t)(i * 3 + 2) * D_MODEL, dn, xd_bf);
    { float* t = dc; dc = dn; dn = t; }
  }

  // ---- Final vocabulary projection: [2048, 512] x [512, 32000] ----
  gemm_bf16_kernel<false, true, false>
      <<<dim3(VOCT / 64, mtiles), 256, 0, stream>>>(
          xd_bf, w_fc, fc_b, out, nullptr, (int)M, VOCT, D_MODEL);
}
